// SelfAttention_11965778887070
// MI455X (gfx1250) — matrix operand order repaired and checked by reading, not compiled, not run on hardware
//
#include <hip/hip_runtime.h>
#include <hip/hip_bf16.h>

// SAGAN-style self-attention for MI455X / gfx1250, wave32 + WMMA bf16.
// B=16, C=256, R=32, CH=128 -> n=4096 queries, m=1024 pooled keys.
//
//  1) repack_w:   weights -> bf16, pre-swizzled into exact A-fragment order
//  2) pool:       xp_bf[b][c][m] = width-avgpool4(x), bf16
//  3) proj_theta: theta[b][n][r] = w_theta @ x      (bf16, A-frag row layout)
//  4) proj_phig:  phi[b][m][r], g_t[b][c][m]        (bf16, B-frag row layout)
//  5) attn:       flash softmax over m per 16-query tile; phi key blocks are
//                 double-buffered into LDS with global_load_async_to_lds_b128
//                 (ASYNCcnt); fused w_o projection + gamma residual.
//                 All GEMMs: v_wmma_f32_16x16x32_bf16.

#define C_   256
#define R_   32
#define CH_  128
#define B_   16
#define N_   4096
#define M_   1024

typedef __attribute__((ext_vector_type(16))) __bf16 v16bf;
typedef __attribute__((ext_vector_type(8)))  __bf16 v8bf;
typedef __attribute__((ext_vector_type(8)))  float  v8f;

union BF16Frag { v16bf v; v8bf h[2]; unsigned short u[16]; };

__device__ __forceinline__ unsigned short f2bf(float f) {
  return __builtin_bit_cast(unsigned short, static_cast<__bf16>(f));
}

__device__ __forceinline__ float rowmax16(float v) {
#pragma unroll
  for (int m = 8; m >= 1; m >>= 1) v = fmaxf(v, __shfl_xor(v, m, 16));
  return v;
}
__device__ __forceinline__ float rowsum16(float v) {
#pragma unroll
  for (int m = 8; m >= 1; m >>= 1) v += __shfl_xor(v, m, 16);
  return v;
}

// Issue one 2 KB contiguous block (32 phi rows) into LDS: 4 x b128 per lane.
__device__ __forceinline__ void async_copy_2k(const void* src, unsigned lds_byte,
                                              int lane) {
#pragma unroll
  for (int k = 0; k < 4; ++k) {
    unsigned ldsa = lds_byte + (unsigned)lane * 16u + (unsigned)k * 512u;
    unsigned long long ga =
        (unsigned long long)src + (unsigned long long)(lane * 16 + k * 512);
    asm volatile("global_load_async_to_lds_b128 %0, %1, off"
                 :: "v"(ldsa), "v"(ga) : "memory");
  }
}

// wswz layout (ushort units): WT @0 (2x8 frags), WP @8192, WG @16384 (8x8),
// WO @49152 (16x4). Fragment (t,q): wswz + ((t*chunks+q)*32 + lane)*16.

// ------------------------------------------------ weight repack into A-frags
__global__ void repack_w_kernel(const float* __restrict__ wt,
                                const float* __restrict__ wp,
                                const float* __restrict__ wg,
                                const float* __restrict__ wo,
                                unsigned short* __restrict__ wswz) {
  int i = blockIdx.x * blockDim.x + threadIdx.x;   // 81920 fragment elements
  const float* src; int cols, chunks, idx;
  if (i < 8192)       { src = wt; cols = 256; chunks = 8; idx = i; }
  else if (i < 16384) { src = wp; cols = 256; chunks = 8; idx = i - 8192; }
  else if (i < 49152) { src = wg; cols = 256; chunks = 8; idx = i - 16384; }
  else                { src = wo; cols = 128; chunks = 4; idx = i - 49152; }
  int e    = idx & 15;
  int lane = (idx >> 4) & 31;
  int tq   = idx >> 9;
  int q    = tq % chunks;
  int t    = tq / chunks;
  int half = lane >> 4, lh = lane & 15;
  int row  = t * 16 + lh;
  int col  = q * 32 + half * 16 + e;                         // {0..15}
  wswz[i] = f2bf(src[(size_t)row * cols + col]);
}

// ----------------------------------------------------- pool (1,4) + cvt bf16
__global__ void pool_kernel(const float* __restrict__ x,
                            unsigned short* __restrict__ xp_bf) {
  int i = blockIdx.x * blockDim.x + threadIdx.x;   // over B*C*64*16
  const float* src = x + ((size_t)(i >> 4) << 6) + ((i & 15) << 2);
  xp_bf[i] = f2bf((src[0] + src[1] + src[2] + src[3]) * 0.25f);
}

// ------------------------------------------------------- theta = w_theta @ x
__global__ __launch_bounds__(128) void proj_theta_kernel(
    const float* __restrict__ x, const unsigned short* __restrict__ WT,
    unsigned short* __restrict__ theta) {
  int wave = blockIdx.x * (blockDim.x >> 5) + (threadIdx.x >> 5);
  int lane = threadIdx.x & 31;
  int half = lane >> 4, lh = lane & 15;
  int b  = wave >> 8;
  int n0 = (wave & 255) << 4;

  v8f zero = {0,0,0,0,0,0,0,0};
  v8f acc0 = zero, acc1 = zero;
#pragma unroll
  for (int q = 0; q < 8; ++q) {                    // K = 256, 8 chunks of 32
    BF16Frag bx;                                   // B[k=c][n] from x[b][c][n]
    const float* xb = x + ((size_t)b * C_ + q * 32 + half * 16) * (size_t)N_ + n0 + lh;
#pragma unroll
    for (int kk = 0; kk < 16; ++kk) bx.u[kk] = f2bf(xb[(size_t)kk * N_]);
#pragma unroll
    for (int rt = 0; rt < 2; ++rt) {
      BF16Frag aw;
      aw.v = *(const v16bf*)(WT + ((size_t)(rt * 8 + q) * 32 + lane) * 16);
      if (rt == 0)
        acc0 = __builtin_amdgcn_wmma_f32_16x16x32_bf16(false, aw.v, false, bx.v, (short)0, acc0, false, false);
      else
        acc1 = __builtin_amdgcn_wmma_f32_16x16x32_bf16(false, aw.v, false, bx.v, (short)0, acc1, false, false);
    }
  }
  unsigned short* tb = theta + ((size_t)b * N_ + n0 + lh) * R_;
#pragma unroll
  for (int j = 0; j < 8; ++j) {
    tb[j + 8 * half]      = f2bf(acc0[j]);
    tb[16 + j + 8 * half] = f2bf(acc1[j]);
  }
}

// --------------------------------------------- phi = w_phi@xp, g = w_g@xp
__global__ __launch_bounds__(128) void proj_phig_kernel(
    const unsigned short* __restrict__ xp_bf,
    const unsigned short* __restrict__ WP, const unsigned short* __restrict__ WG,
    unsigned short* __restrict__ phi, unsigned short* __restrict__ g_t) {
  int wave = blockIdx.x * (blockDim.x >> 5) + (threadIdx.x >> 5);
  int lane = threadIdx.x & 31;
  int half = lane >> 4, lh = lane & 15;
  int b  = wave >> 6;
  int m0 = (wave & 63) << 4;

  v8f zero = {0,0,0,0,0,0,0,0};
  v8f accP0 = zero, accP1 = zero;
  v8f accG[8];
#pragma unroll
  for (int f = 0; f < 8; ++f) accG[f] = zero;

#pragma unroll
  for (int q = 0; q < 8; ++q) {
    BF16Frag bx;                                   // B[k=c][n=m] from xp_bf
    const unsigned short* xb =
        xp_bf + ((size_t)b * C_ + q * 32 + half * 16) * (size_t)M_ + m0 + lh;
#pragma unroll
    for (int kk = 0; kk < 16; ++kk) bx.u[kk] = xb[(size_t)kk * M_];
#pragma unroll
    for (int rt = 0; rt < 2; ++rt) {
      BF16Frag aw;
      aw.v = *(const v16bf*)(WP + ((size_t)(rt * 8 + q) * 32 + lane) * 16);
      if (rt == 0)
        accP0 = __builtin_amdgcn_wmma_f32_16x16x32_bf16(false, aw.v, false, bx.v, (short)0, accP0, false, false);
      else
        accP1 = __builtin_amdgcn_wmma_f32_16x16x32_bf16(false, aw.v, false, bx.v, (short)0, accP1, false, false);
    }
#pragma unroll
    for (int ct = 0; ct < 8; ++ct) {
      BF16Frag aw;
      aw.v = *(const v16bf*)(WG + ((size_t)(ct * 8 + q) * 32 + lane) * 16);
      accG[ct] = __builtin_amdgcn_wmma_f32_16x16x32_bf16(false, aw.v, false, bx.v, (short)0, accG[ct], false, false);
    }
  }
  unsigned short* pb = phi + ((size_t)b * M_ + m0 + lh) * R_;
#pragma unroll
  for (int j = 0; j < 8; ++j) {
    pb[j + 8 * half]      = f2bf(accP0[j]);
    pb[16 + j + 8 * half] = f2bf(accP1[j]);
  }
#pragma unroll
  for (int ct = 0; ct < 8; ++ct)
#pragma unroll
    for (int j = 0; j < 8; ++j) {
      int c = ct * 16 + j + 8 * half;
      g_t[((size_t)b * CH_ + c) * M_ + m0 + lh] = f2bf(accG[ct][j]);
    }
}

// ------------------------------------------- flash attention + out proj
// Per-wave LDS (ushort units): pbuf 512 | obuf 2048 | phi staging 2 x 1024.
#define WV_USH  4608
#define WV_B    9216

__global__ __launch_bounds__(128) void attn_kernel(
    const float* __restrict__ x, const unsigned short* __restrict__ theta,
    const unsigned short* __restrict__ phi, const unsigned short* __restrict__ g_t,
    const unsigned short* __restrict__ WO, const float* __restrict__ gamma,
    float* __restrict__ out) {
  __shared__ __align__(32) unsigned short smem[4 * WV_USH];   // LDS offset 0
  int wib = threadIdx.x >> 5;
  unsigned short* pbuf = smem + wib * WV_USH;      // P bf16 [n_local][m_local] 16x32
  unsigned short* obuf = pbuf + 512;               // O bf16 [n_local][c]      16x128
  unsigned short* phis = obuf + 2048;              // phi staging, 2 x 2 KB
  unsigned phis_byte = (unsigned)wib * WV_B + (512u + 2048u) * 2u;
  int wave = blockIdx.x * 4 + wib;
  int lane = threadIdx.x & 31;
  int half = lane >> 4, lh = lane & 15;
  int b  = wave >> 8;
  int n0 = (wave & 255) << 4;

  v8f zero = {0,0,0,0,0,0,0,0};

  BF16Frag aTh;                                    // A[n][r], K = R_ = 32
  {
    const unsigned short* tr = theta + ((size_t)b * N_ + n0 + lh) * R_ + half * 16;
    aTh.v = *(const v16bf*)tr;
  }

  float run_max[8], run_sum[8];
  v8f O[8];
#pragma unroll
  for (int j = 0; j < 8; ++j) { run_max[j] = -3.0e38f; run_sum[j] = 0.0f; }
#pragma unroll
  for (int f = 0; f < 8; ++f) O[f] = zero;

  // prologue: stage chunk 0 of phi (32 rows x 64 B, contiguous) into buffer 0
  async_copy_2k(phi + ((size_t)b * M_) * R_, phis_byte, lane);

  for (int mc = 0; mc < M_; mc += 32) {            // stream keys, 32 per step
    int cur = (mc >> 5) & 1;
    asm volatile("s_wait_asynccnt 0x0" ::: "memory");   // cur phi block resident
    if (mc + 32 < M_)                               // overlap next block copy
      async_copy_2k(phi + ((size_t)b * M_ + mc + 32) * R_,
                    phis_byte + (unsigned)(cur ^ 1) * 2048u, lane);
    __builtin_prefetch(g_t + ((size_t)b * CH_ + lh) * M_ + mc, 0, 0);

    const unsigned short* pc = phis + cur * 1024;  // staged [row][r] bf16
    BF16Frag b0, b1;                               // B[k=r][n=m]
    b0.v = *(const v16bf*)(pc + lh * 32 + half * 16);
    b1.v = *(const v16bf*)(pc + (16 + lh) * 32 + half * 16);
    v8f S0 = __builtin_amdgcn_wmma_f32_16x16x32_bf16(false, aTh.v, false, b0.v, (short)0, zero, false, false);
    v8f S1 = __builtin_amdgcn_wmma_f32_16x16x32_bf16(false, aTh.v, false, b1.v, (short)0, zero, false, false);

    float nm[8], sc[8], P0[8], P1[8];
#pragma unroll
    for (int j = 0; j < 8; ++j) {
      float mx = rowmax16(fmaxf(S0[j], S1[j]));
      nm[j] = fmaxf(run_max[j], mx);
      sc[j] = __expf(run_max[j] - nm[j]);
      run_max[j] = nm[j];
    }
#pragma unroll
    for (int j = 0; j < 8; ++j) {
      P0[j] = __expf(S0[j] - nm[j]);
      P1[j] = __expf(S1[j] - nm[j]);
      run_sum[j] = run_sum[j] * sc[j] + rowsum16(P0[j] + P1[j]);
    }
#pragma unroll
    for (int f = 0; f < 8; ++f)
#pragma unroll
      for (int j = 0; j < 8; ++j) O[f][j] *= sc[j];

    // C-layout -> A-layout transpose of P through LDS (bf16, intra-wave)
#pragma unroll
    for (int j = 0; j < 8; ++j) {
      pbuf[(j + 8 * half) * 32 + lh]      = f2bf(P0[j]);
      pbuf[(j + 8 * half) * 32 + 16 + lh] = f2bf(P1[j]);
    }
    asm volatile("s_wait_dscnt 0" ::: "memory");
    BF16Frag aP;
    aP.v = *(const v16bf*)(pbuf + lh * 32 + half * 16);
#pragma unroll
    for (int f = 0; f < 8; ++f) {                  // O[n][c] += P[n][m] g_t[c][m]
      BF16Frag bg;
      bg.v = *(const v16bf*)(g_t + ((size_t)b * CH_ + f * 16 + lh) * M_ + mc + half * 16);
      O[f] = __builtin_amdgcn_wmma_f32_16x16x32_bf16(false, aP.v, false, bg.v, (short)0, O[f], false, false);
    }
  }

  // normalize; park O as bf16 [n][c] so B-frag rows are contiguous in LDS
  float rinv[8];
#pragma unroll
  for (int j = 0; j < 8; ++j) rinv[j] = 1.0f / run_sum[j];
#pragma unroll
  for (int f = 0; f < 8; ++f)
#pragma unroll
    for (int j = 0; j < 8; ++j)
      obuf[(j + 8 * half) * 128 + f * 16 + lh] = f2bf(O[f][j] * rinv[j]);
  asm volatile("s_wait_dscnt 0" ::: "memory");

  BF16Frag bo[4];                                  // B[k=c][n] for w_o GEMM
#pragma unroll
  for (int q = 0; q < 4; ++q)
    bo[q].v = *(const v16bf*)(obuf + lh * 128 + q * 32 + half * 16);

  float gm = gamma[0];
#pragma unroll 1
  for (int t = 0; t < 16; ++t) {                   // C_ = 256 -> 16 o-tiles
    v8f acc = zero;
#pragma unroll
    for (int q = 0; q < 4; ++q) {
      BF16Frag aw;                                 // pre-swizzled w_o A-frag
      aw.v = *(const v16bf*)(WO + ((size_t)(t * 4 + q) * 32 + lane) * 16);
      acc = __builtin_amdgcn_wmma_f32_16x16x32_bf16(false, aw.v, false, bo[q].v, (short)0, acc, false, false);
    }
#pragma unroll
    for (int j = 0; j < 8; ++j) {
      int o = t * 16 + j + 8 * half;
      size_t idx = ((size_t)b * C_ + o) * (size_t)N_ + n0 + lh;
      out[idx] = x[idx] + gm * acc[j];
    }
  }
}

// ---------------------------------------------------------------- launcher
extern "C" void kernel_launch(void* const* d_in, const int* in_sizes, int n_in,
                              void* d_out, int out_size, void* d_ws, size_t ws_size,
                              hipStream_t stream) {
  const float* x       = (const float*)d_in[0];
  const float* w_theta = (const float*)d_in[1];
  const float* w_phi   = (const float*)d_in[2];
  const float* w_g     = (const float*)d_in[3];
  const float* w_o     = (const float*)d_in[4];
  const float* gamma   = (const float*)d_in[5];
  float* out = (float*)d_out;

  char* ws = (char*)d_ws;                                       // ~17.2 MiB used
  unsigned short* xp_bf = (unsigned short*)ws;                  // 8 MiB
  unsigned short* theta = (unsigned short*)(ws + 8388608);      // 4 MiB
  unsigned short* phi   = (unsigned short*)(ws + 12582912);     // 1 MiB
  unsigned short* g_t   = (unsigned short*)(ws + 13631488);     // 4 MiB
  unsigned short* wswz  = (unsigned short*)(ws + 17825792);     // 160 KiB
  unsigned short* WT = wswz;
  unsigned short* WP = wswz + 8192;
  unsigned short* WG = wswz + 16384;
  unsigned short* WO = wswz + 49152;

  repack_w_kernel<<<320, 256, 0, stream>>>(w_theta, w_phi, w_g, w_o, wswz);
  pool_kernel<<<16384, 256, 0, stream>>>(x, xp_bf);
  proj_theta_kernel<<<1024, 128, 0, stream>>>(x, WT, theta);
  proj_phig_kernel<<<256, 128, 0, stream>>>(xp_bf, WP, WG, phi, g_t);
  attn_kernel<<<1024, 128, 0, stream>>>(x, theta, phi, g_t, WO, gamma, out);
}
